// ATULayer_76304388981138
// MI455X (gfx1250) — matrix operand order repaired and checked by reading, not compiled, not run on hardware
//
#include <hip/hip_runtime.h>

// ---------------- problem constants ----------------
#define NMB   128          // N*M batches
#define TT    256          // tokens
#define DIM   960          // C*V
#define INNER 1024         // HEADS*DIM_HEAD
#define NHEAD 8
#define DHEAD 128
#define LNEPS 1e-5f

#define ASG __attribute__((address_space(1)))   // global
#define ASL __attribute__((address_space(3)))   // LDS

typedef __attribute__((ext_vector_type(16))) __bf16        v16bf;
typedef __attribute__((ext_vector_type(8)))  float         v8f;
typedef __attribute__((ext_vector_type(8)))  unsigned int  v8u;
typedef __attribute__((ext_vector_type(4)))  int           v4i;

__device__ __forceinline__ unsigned short f2bf(float f) {
  unsigned u = __float_as_uint(f);
  u += 0x7FFFu + ((u >> 16) & 1u);   // round-to-nearest-even
  return (unsigned short)(u >> 16);
}
__device__ __forceinline__ float bf2f(unsigned short h) {
  return __uint_as_float(((unsigned)h) << 16);
}

__device__ __forceinline__ void async_copy16(const unsigned short* g,
                                             unsigned short* l) {
  // global -> LDS, 16 bytes per lane, tracked by ASYNCcnt (no VGPR staging)
  __builtin_amdgcn_global_load_async_to_lds_b128(
      (ASG v4i*)g, (ASL v4i*)l, 0, 0);
}
__device__ __forceinline__ void wait_async0() {
  asm volatile("s_wait_asynccnt 0x0" ::: "memory");
}

// ---------------- small helper kernels ----------------
__global__ void cvt_bf16_kernel(const float* __restrict__ src,
                                unsigned short* __restrict__ dst, int n) {
  int i = blockIdx.x * 256 + threadIdx.x;
  if (i < n) dst[i] = f2bf(src[i]);
}

// One block per nm; thread = t; loop cv (lanes along t -> coalesced).
__global__ void ln_stats_kernel(const float* __restrict__ x,
                                float* __restrict__ mu, float* __restrict__ rstd) {
  int nm = blockIdx.x;
  int t  = threadIdx.x;
  const float* p = x + (size_t)nm * DIM * TT + t;
  float s = 0.f, ss = 0.f;
  for (int cv = 0; cv < DIM; ++cv) {
    float v = p[(size_t)cv * TT];
    s += v; ss += v * v;
  }
  float m   = s * (1.f / DIM);
  float var = ss * (1.f / DIM) - m * m;
  mu[nm * TT + t]   = m;
  rstd[nm * TT + t] = rsqrtf(var + LNEPS);
}

// Normalize + transpose [nm,cv,t] -> bf16 [nm,t,cv] via 32x32 LDS tile.
__global__ void ln_apply_kernel(const float* __restrict__ x,
                                const float* __restrict__ mu,
                                const float* __restrict__ rstd,
                                const float* __restrict__ g,
                                const float* __restrict__ b,
                                unsigned short* __restrict__ xn) {
  __shared__ float tile[32][33];
  int nm = blockIdx.x;
  int t0 = blockIdx.y * 32;
  int c0 = blockIdx.z * 32;
  int tid = threadIdx.x;
#pragma unroll
  for (int i = 0; i < 4; ++i) {
    int e = tid + i * 256, r = e >> 5, c = e & 31;   // r=cv_local, c=t_local
    tile[r][c] = x[((size_t)nm * DIM + c0 + r) * TT + t0 + c];
  }
  __syncthreads();
#pragma unroll
  for (int i = 0; i < 4; ++i) {
    int e = tid + i * 256, r = e >> 5, c = e & 31;   // r=t_local, c=cv_local
    int t = t0 + r, cv = c0 + c;
    float m  = mu[nm * TT + t];
    float rs = rstd[nm * TT + t];
    float v  = (tile[c][r] - m) * rs * g[cv] + b[cv];
    xn[((size_t)nm * TT + t) * DIM + cv] = f2bf(v);
  }
}

// In-place softmax on bf16 rows of length 256 (scale already applied).
__global__ void softmax_kernel(unsigned short* __restrict__ S) {
  __shared__ float red[256];
  int tid = threadIdx.x;
  unsigned short* p = S + (size_t)blockIdx.x * TT;
  float v = bf2f(p[tid]);
  red[tid] = v; __syncthreads();
  for (int s = 128; s > 0; s >>= 1) {
    if (tid < s) red[tid] = fmaxf(red[tid], red[tid + s]);
    __syncthreads();
  }
  float mx = red[0]; __syncthreads();
  float e = __expf(v - mx);
  red[tid] = e; __syncthreads();
  for (int s = 128; s > 0; s >>= 1) {
    if (tid < s) red[tid] += red[tid + s];
    __syncthreads();
  }
  p[tid] = f2bf(e / red[0]);
}

// ---------------- shared WMMA bf16 GEMM mainloop ----------------
// BM=BN=128, BK=32, 256 threads = 8 waves (4 over M x 2 over N).
// Each wave: 2x4 tiles of 16x16 -> v8f acc[2][4].
// Double-buffered LDS; A tile (and B tile when BT) moved with
// GLOBAL_LOAD_ASYNC_TO_LDS_B128 (ASYNCcnt), prefetching tile k+1 under the
// WMMAs of tile k.
// A: row-major [M,K]. BT=true : B row-major [N,K] (straight into [n][k] LDS).
//                    BT=false: B row-major [K,N] (VGPR-staged transpose; N-guard).
#define BM 128
#define BN 128
#define BK 32
#define ATILE (BM * BK)
#define BTILE (BN * BK)

template<bool BT>
__device__ __forceinline__ void gemm_block(
    const unsigned short* __restrict__ A, int lda,
    const unsigned short* __restrict__ B, int ldb,
    int K, int n0, int Ntot,
    unsigned short* As, unsigned short* Bs, v8f (*acc)[4]) {
  const int tid  = threadIdx.x;
  const int wid  = tid >> 5, lane = tid & 31;
  const int half = lane >> 4, l16 = lane & 15;
  const int wm = (wid & 3) * 32, wn = (wid >> 2) * 64;
  const int nk = K / BK;

  auto load_tiles = [&](int k0, int buf) {
    unsigned short* Ab = As + buf * ATILE;
    unsigned short* Bb = Bs + buf * BTILE;
#pragma unroll
    for (int i = 0; i < 2; ++i) {
      int chunk = tid + i * 256;
      int r = chunk >> 2, c8 = (chunk & 3) * 8;   // row, 8-elem col chunk
      async_copy16(A + (size_t)r * lda + k0 + c8, Ab + r * BK + c8);
    }
    if (BT) {
#pragma unroll
      for (int i = 0; i < 2; ++i) {
        int chunk = tid + i * 256;
        int r = chunk >> 2, c8 = (chunk & 3) * 8;
        async_copy16(B + (size_t)r * ldb + k0 + c8, Bb + r * BK + c8);
      }
    } else {
#pragma unroll
      for (int i = 0; i < 2; ++i) {
        int chunk = tid + i * 256;
        int r = chunk >> 4, c8 = (chunk & 15) * 8;        // r = k row, c8 = n col
        const unsigned short* src = B + (size_t)(k0 + r) * ldb + c8;
        unsigned short tmp[8];
        if (n0 + c8 + 8 <= Ntot) {
          *(uint4*)tmp = *(const uint4*)src;
        } else {
#pragma unroll
          for (int j = 0; j < 8; ++j) tmp[j] = (n0 + c8 + j < Ntot) ? src[j] : 0;
        }
#pragma unroll
        for (int j = 0; j < 8; ++j) Bb[(c8 + j) * BK + r] = tmp[j];  // transpose
      }
    }
  };

  load_tiles(0, 0);
  wait_async0();
  __syncthreads();

  for (int ki = 0; ki < nk; ++ki) {
    const int cur = ki & 1;
    if (ki + 1 < nk) load_tiles((ki + 1) * BK, cur ^ 1);  // prefetch under compute

    const unsigned short* Ab = As + cur * ATILE;
    const unsigned short* Bb = Bs + cur * BTILE;

    v16bf af[2], bfr[4];
#pragma unroll
    for (int mi = 0; mi < 2; ++mi) {
      const unsigned short* s = Ab + (wm + mi * 16 + l16) * BK + half * 8;
      uint4 lo = *(const uint4*)s;
      uint4 hi = *(const uint4*)(s + 16);
      v8u p;
      p[0]=lo.x; p[1]=lo.y; p[2]=lo.z; p[3]=lo.w;
      p[4]=hi.x; p[5]=hi.y; p[6]=hi.z; p[7]=hi.w;
      af[mi] = __builtin_bit_cast(v16bf, p);
    }
#pragma unroll
    for (int ni = 0; ni < 4; ++ni) {
      const unsigned short* s = Bb + (wn + ni * 16 + l16) * BK + half * 8;
      uint4 lo = *(const uint4*)s;
      uint4 hi = *(const uint4*)(s + 16);
      v8u p;
      p[0]=lo.x; p[1]=lo.y; p[2]=lo.z; p[3]=lo.w;
      p[4]=hi.x; p[5]=hi.y; p[6]=hi.z; p[7]=hi.w;
      bfr[ni] = __builtin_bit_cast(v16bf, p);
    }
#pragma unroll
    for (int mi = 0; mi < 2; ++mi)
#pragma unroll
      for (int ni = 0; ni < 4; ++ni)
        acc[mi][ni] = __builtin_amdgcn_wmma_f32_16x16x32_bf16(
            false, af[mi], false, bfr[ni], (short)0, acc[mi][ni], false, false);

    wait_async0();       // prefetched tile resident before next iteration reads it
    __syncthreads();
  }
}

#define GEMM_PROLOGUE()                                               \
  __shared__ __align__(16) unsigned short As[2 * ATILE];              \
  __shared__ __align__(16) unsigned short Bs[2 * BTILE];              \
  v8f acc[2][4];                                                      \
  for (int mi = 0; mi < 2; ++mi)                                      \
    for (int ni = 0; ni < 4; ++ni)                                    \
      for (int j = 0; j < 8; ++j) acc[mi][ni][j] = 0.f;

#define EPILOGUE_IDX()                                                \
  const int tid = threadIdx.x;                                        \
  const int wid = tid >> 5, lane = tid & 31;                          \
  const int half = lane >> 4, l16 = lane & 15;                        \
  const int wm = (wid & 3) * 32, wn = (wid >> 2) * 64;

// ---- QKV: [32768,960] x [960,3072] -> scatter bf16 q/k/v [nm,h,t,d] ----
__global__ void __launch_bounds__(256) gemm_qkv_kernel(
    const unsigned short* __restrict__ xn, const unsigned short* __restrict__ wq,
    unsigned short* __restrict__ dq, unsigned short* __restrict__ dk,
    unsigned short* __restrict__ dv) {
  GEMM_PROLOGUE();
  int m0 = blockIdx.x * BM, n0 = blockIdx.y * BN;
  gemm_block<false>(xn + (size_t)m0 * DIM, DIM, wq + n0, 3 * INNER,
                    DIM, n0, 3 * INNER, As, Bs, acc);
  EPILOGUE_IDX();
#pragma unroll
  for (int mi = 0; mi < 2; ++mi)
#pragma unroll
    for (int ni = 0; ni < 4; ++ni)
#pragma unroll
      for (int r = 0; r < 8; ++r) {
        int grow = m0 + wm + mi * 16 + half * 8 + r;
        int gcol = n0 + wn + ni * 16 + l16;
        int nm = grow >> 8, t = grow & 255;
        int part = gcol >> 10, h = (gcol >> 7) & 7, d = gcol & 127;
        unsigned short* dst = (part == 0) ? dq : (part == 1) ? dk : dv;
        dst[(((size_t)nm * NHEAD + h) * TT + t) * DHEAD + d] = f2bf(acc[mi][ni][r]);
      }
}

// ---- S = scale * Q Kt : batched (z = nm*8+h), NT ----
__global__ void __launch_bounds__(256) gemm_s_kernel(
    const unsigned short* __restrict__ q, const unsigned short* __restrict__ k,
    unsigned short* __restrict__ S) {
  GEMM_PROLOGUE();
  int z = blockIdx.z;
  int m0 = blockIdx.x * BM, n0 = blockIdx.y * BN;
  const unsigned short* A = q + (size_t)z * TT * DHEAD + (size_t)m0 * DHEAD;
  const unsigned short* B = k + (size_t)z * TT * DHEAD + (size_t)n0 * DHEAD;
  gemm_block<true>(A, DHEAD, B, DHEAD, DHEAD, n0, TT, As, Bs, acc);
  EPILOGUE_IDX();
  const float scale = 0.08838834764831845f;  // 128^-0.5
  unsigned short* out = S + (size_t)z * TT * TT;
#pragma unroll
  for (int mi = 0; mi < 2; ++mi)
#pragma unroll
    for (int ni = 0; ni < 4; ++ni)
#pragma unroll
      for (int r = 0; r < 8; ++r) {
        int grow = m0 + wm + mi * 16 + half * 8 + r;
        int gcol = n0 + wn + ni * 16 + l16;
        out[(size_t)grow * TT + gcol] = f2bf(acc[mi][ni][r] * scale);
      }
}

// ---- O = P V : batched, NN -> attn_out [nm,t,h*128+d] bf16 ----
__global__ void __launch_bounds__(256) gemm_av_kernel(
    const unsigned short* __restrict__ P, const unsigned short* __restrict__ v,
    unsigned short* __restrict__ ao) {
  GEMM_PROLOGUE();
  int z = blockIdx.z;
  int m0 = blockIdx.x * BM;
  const unsigned short* A = P + (size_t)z * TT * TT + (size_t)m0 * TT;
  const unsigned short* B = v + (size_t)z * TT * DHEAD;
  gemm_block<false>(A, TT, B, DHEAD, TT, 0, DHEAD, As, Bs, acc);
  EPILOGUE_IDX();
  int nm = z >> 3, h = z & 7;
#pragma unroll
  for (int mi = 0; mi < 2; ++mi)
#pragma unroll
    for (int ni = 0; ni < 4; ++ni)
#pragma unroll
      for (int r = 0; r < 8; ++r) {
        int grow = m0 + wm + mi * 16 + half * 8 + r;     // t
        int gcol = wn + ni * 16 + l16;                   // d (0..127)
        ao[((size_t)nm * TT + grow) * INNER + h * DHEAD + gcol] = f2bf(acc[mi][ni][r]);
      }
}

// ---- out-proj + bias + residual(x) -> r_t [nm,cv,t] bf16 ----
__global__ void __launch_bounds__(256) gemm_outproj_kernel(
    const unsigned short* __restrict__ ao, const unsigned short* __restrict__ wo,
    const float* __restrict__ b_out, const float* __restrict__ x,
    unsigned short* __restrict__ rt) {
  GEMM_PROLOGUE();
  int m0 = blockIdx.x * BM, n0 = blockIdx.y * BN;
  gemm_block<false>(ao + (size_t)m0 * INNER, INNER, wo + n0, DIM,
                    INNER, n0, DIM, As, Bs, acc);
  EPILOGUE_IDX();
#pragma unroll
  for (int mi = 0; mi < 2; ++mi)
#pragma unroll
    for (int ni = 0; ni < 4; ++ni)
#pragma unroll
      for (int r = 0; r < 8; ++r) {
        int grow = m0 + wm + mi * 16 + half * 8 + r;
        int gcol = n0 + wn + ni * 16 + l16;              // cv
        if (gcol < DIM) {
          int nm = grow >> 8, t = grow & 255;
          size_t xi = ((size_t)nm * DIM + gcol) * TT + t;  // x layout == [nm,cv,t]
          float val = acc[mi][ni][r] + b_out[gcol] + x[xi];
          rt[xi] = f2bf(val);
        }
      }
}

// ---- final: tanh(r_t [122880,256] x w_lin^T + b_lin) -> fp32 out ----
__global__ void __launch_bounds__(256) gemm_final_kernel(
    const unsigned short* __restrict__ rt, const unsigned short* __restrict__ wl,
    const float* __restrict__ b_lin, float* __restrict__ out) {
  GEMM_PROLOGUE();
  int m0 = blockIdx.x * BM, n0 = blockIdx.y * BN;
  gemm_block<true>(rt + (size_t)m0 * TT, TT, wl + (size_t)n0 * TT, TT,
                   TT, n0, TT, As, Bs, acc);
  EPILOGUE_IDX();
#pragma unroll
  for (int mi = 0; mi < 2; ++mi)
#pragma unroll
    for (int ni = 0; ni < 4; ++ni)
#pragma unroll
      for (int r = 0; r < 8; ++r) {
        int grow = m0 + wm + mi * 16 + half * 8 + r;
        int gcol = n0 + wn + ni * 16 + l16;
        out[(size_t)grow * TT + gcol] = tanhf(acc[mi][ni][r] + b_lin[gcol]);
      }
}

// ---------------- host launcher ----------------
extern "C" void kernel_launch(void* const* d_in, const int* in_sizes, int n_in,
                              void* d_out, int out_size, void* d_ws, size_t ws_size,
                              hipStream_t stream) {
  const float* x     = (const float*)d_in[0];
  const float* ln_g  = (const float*)d_in[1];
  const float* ln_b  = (const float*)d_in[2];
  const float* w_qkv = (const float*)d_in[3];
  const float* w_out = (const float*)d_in[4];
  const float* b_out = (const float*)d_in[5];
  const float* w_lin = (const float*)d_in[6];
  const float* b_lin = (const float*)d_in[7];
  float* out = (float*)d_out;

  char* ws = (char*)d_ws;
  size_t off = 0;
  auto alloc = [&](size_t bytes) -> char* {
    char* p = ws + off;
    off += (bytes + 255) & ~(size_t)255;
    return p;
  };
  unsigned short* xn   = (unsigned short*)alloc((size_t)NMB * TT * DIM * 2);
  unsigned short* wqkv = (unsigned short*)alloc((size_t)DIM * 3 * INNER * 2);
  unsigned short* wout = (unsigned short*)alloc((size_t)INNER * DIM * 2);
  unsigned short* wlin = (unsigned short*)alloc((size_t)TT * TT * 2);
  float*          mu   = (float*)alloc((size_t)NMB * TT * 4);
  float*          rstd = (float*)alloc((size_t)NMB * TT * 4);
  unsigned short* dq   = (unsigned short*)alloc((size_t)NMB * NHEAD * TT * DHEAD * 2);
  unsigned short* dk   = (unsigned short*)alloc((size_t)NMB * NHEAD * TT * DHEAD * 2);
  unsigned short* dv   = (unsigned short*)alloc((size_t)NMB * NHEAD * TT * DHEAD * 2);
  unsigned short* S    = (unsigned short*)alloc((size_t)NMB * NHEAD * TT * TT * 2);
  unsigned short* ao   = (unsigned short*)alloc((size_t)NMB * TT * INNER * 2);
  unsigned short* rt   = (unsigned short*)alloc((size_t)NMB * DIM * TT * 2);

  int nqkv = DIM * 3 * INNER, nout = INNER * DIM, nlin = TT * TT;
  cvt_bf16_kernel<<<(nqkv + 255) / 256, 256, 0, stream>>>(w_qkv, wqkv, nqkv);
  cvt_bf16_kernel<<<(nout + 255) / 256, 256, 0, stream>>>(w_out, wout, nout);
  cvt_bf16_kernel<<<(nlin + 255) / 256, 256, 0, stream>>>(w_lin, wlin, nlin);

  ln_stats_kernel<<<NMB, 256, 0, stream>>>(x, mu, rstd);
  ln_apply_kernel<<<dim3(NMB, TT / 32, DIM / 32), 256, 0, stream>>>(
      x, mu, rstd, ln_g, ln_b, xn);

  gemm_qkv_kernel<<<dim3((NMB * TT) / BM, (3 * INNER) / BN), 256, 0, stream>>>(
      xn, wqkv, dq, dk, dv);
  gemm_s_kernel<<<dim3(TT / BM, TT / BN, NMB * NHEAD), 256, 0, stream>>>(dq, dk, S);
  softmax_kernel<<<NMB * NHEAD * TT, 256, 0, stream>>>(S);
  gemm_av_kernel<<<dim3(TT / BM, 1, NMB * NHEAD), 256, 0, stream>>>(S, dv, ao);
  gemm_outproj_kernel<<<dim3((NMB * TT) / BM, (DIM + BN - 1) / BN), 256, 0, stream>>>(
      ao, wout, b_out, x, rt);
  gemm_final_kernel<<<dim3((NMB * DIM) / BM, TT / BN), 256, 0, stream>>>(
      rt, wlin, b_lin, out);
}
